// ConvSepKanAttention_60198261621029
// MI455X (gfx1250) — compile-verified, matrix operand order checked by reading, not run on hardware
//
#include <hip/hip_runtime.h>

typedef _Float16 h16;
typedef __attribute__((ext_vector_type(16))) _Float16 v16h;
typedef __attribute__((ext_vector_type(8)))  float    v8f;

#define NB 16
#define C  128
#define HH 128
#define WW 128
#define HW 16384

// Map fragment element e (0..15) + lane-half to K index per CDNA5 ISA 16-bit
// A/B layout: VGPR j (=e/2) holds K pair; j<4 -> K=2j+p(+8*half), j>=4 -> 16+...
__device__ __forceinline__ int kmap(int e, int half) {
  int j = e >> 1;
  return ((j & 4) << 2) | ((j & 3) << 1) | (e & 1) | (half << 3);
}

// Async global->LDS 16-byte copy (CDNA5 GLOBAL_LOAD_ASYNC_TO_LDS_B128, ASYNCcnt)
__device__ __forceinline__ void async_g2l_b128(unsigned lds_off, const void* gaddr) {
  asm volatile("global_load_async_to_lds_b128 %0, %1, off"
               :: "v"(lds_off), "v"(gaddr) : "memory");
}
__device__ __forceinline__ void wait_async0() {
  asm volatile("s_wait_asynccnt 0x0" ::: "memory");
}
__device__ __forceinline__ unsigned lds_off_of(const void* p) {
  return (unsigned)(uintptr_t)p;   // flat LDS aperture: addr[31:0] = LDS offset
}

__global__ void __launch_bounds__(256) zero_f32_kernel(float* __restrict__ p, int n) {
  int i = blockIdx.x * 256 + threadIdx.x;
  if (i < n) p[i] = 0.f;
}

// ---------------- Kernel 0: one-shot weight f32 -> f16 conversion ----------------
__global__ void __launch_bounds__(256) cvt_weights_kernel(
    const float* __restrict__ w0, const float* __restrict__ w1,
    const float* __restrict__ w2, h16* __restrict__ dst)
{
  const int i = blockIdx.x * 256 + threadIdx.x;    // [0, 3*C*C)
  const int which = i / (C * C);
  const int off   = i % (C * C);
  const float* src = (which == 0) ? w0 : (which == 1) ? w1 : w2;
  dst[i] = (h16)src[off];
}

// ---------------- Kernel 1: fused Q/K/V 1x1 projections (WMMA) ----------------
// One block = one (b, ntile); wave id = mtile.
// x tile: async-staged f32 -> converted/transposed once to f16 -> B frags are
// two contiguous 16B LDS reads per WMMA, no cvt in the hot loop.
__global__ void __launch_bounds__(256) qkv_proj_kernel(
    const float* __restrict__ x,
    const h16* __restrict__ w3,                // [3][C*C] f16 weights (q,k,v)
    const float* __restrict__ qbias, const float* __restrict__ kbias,
    const float* __restrict__ vbias,
    h16* __restrict__ q0, h16* __restrict__ k0, h16* __restrict__ v0)
{
  __shared__ float xs[C * 16];                 // 8 KB: x[k][n] f32 (async landing)
  __shared__ h16   xt[16 * C];                 // 4 KB: x[n][k] f16 (transposed)
  const int tid  = threadIdx.x;
  const int lane = tid & 31;
  const int mt   = tid >> 5;                   // wave id = out-channel tile
  const int l16  = lane & 15;
  const int half = lane >> 4;
  const int b    = blockIdx.x >> 10;
  const int nt   = blockIdx.x & 1023;
  const int n0   = nt * 16;
  const float* xb = x + (size_t)b * C * HW;

  // stage: each thread async-copies 32B (8 floats) of the tile
  {
    const int row = tid >> 1;
    const int col = (tid & 1) * 8;
    const unsigned dst = lds_off_of(&xs[row * 16 + col]);
    const float* src = xb + (size_t)row * HW + n0 + col;
    async_g2l_b128(dst, src);
    async_g2l_b128(dst + 16u, src + 4);
    wait_async0();
  }
  __syncthreads();

  // convert + transpose once: thread handles 8 k's of one n column
  {
    const int n  = tid & 15;
    const int k0 = (tid >> 4) * 8;
#pragma unroll
    for (int j = 0; j < 8; ++j)
      xt[n * C + k0 + j] = (h16)xs[(k0 + j) * 16 + n];
  }
  __syncthreads();

  const h16* wq = w3 + (size_t)0 * C * C;
  const h16* wk = w3 + (size_t)1 * C * C;
  const h16* wv = w3 + (size_t)2 * C * C;
  const int m0 = mt * 16 + l16;
  v8f accq = {}; v8f acck = {}; v8f accv = {};
  for (int kb = 0; kb < C; kb += 32) {
    v16h bx, aq, ak, av;
#pragma unroll
    for (int e = 0; e < 16; ++e) {
      const int kk = kb + kmap(e, half);
      bx[e] = xt[l16 * C + kk];                // B: [K=cin][N=spatial] (f16 LDS)
      aq[e] = wq[m0 * C + kk];                 // A: [M=cout][K=cin] (f16 global)
      ak[e] = wk[m0 * C + kk];
      av[e] = wv[m0 * C + kk];
    }
    accq = __builtin_amdgcn_wmma_f32_16x16x32_f16(false, aq, false, bx, (short)0, accq, false, false);
    acck = __builtin_amdgcn_wmma_f32_16x16x32_f16(false, ak, false, bx, (short)0, acck, false, false);
    accv = __builtin_amdgcn_wmma_f32_16x16x32_f16(false, av, false, bx, (short)0, accv, false, false);
  }
#pragma unroll
  for (int i = 0; i < 8; ++i) {
    const int m = mt * 16 + i + half * 8;      // C/D layout: VGPR i -> M = i + 8*half
    const size_t o = ((size_t)b * C + m) * HW + n0 + l16;
    q0[o] = (h16)(accq[i] + qbias[m]);
    k0[o] = (h16)(acck[i] + kbias[m]);
    v0[o] = (h16)(accv[i] + vbias[m]);
  }
}

// ---------------- Kernel 2: depthwise 3x3 + bias (+ sum-of-squares) ----------------
__global__ void __launch_bounds__(256) dwconv_kernel(
    const h16* __restrict__ src, const float* __restrict__ w9,
    const float* __restrict__ bias, h16* __restrict__ dst,
    float* __restrict__ sumsq)
{
  const int id    = blockIdx.x;                // b*128*64 + c*64 + chunk
  const int chunk = id & 63;
  const int c     = (id >> 6) & 127;
  const int b     = id >> 13;
  const int s     = chunk * 256 + threadIdx.x; // spatial index
  const int h     = s >> 7;
  const int wcol  = s & 127;
  const h16* p = src + ((size_t)b * C + c) * HW;

  float acc = bias[c];
#pragma unroll
  for (int dh = -1; dh <= 1; ++dh)
#pragma unroll
    for (int dw = -1; dw <= 1; ++dw) {
      const int hh = h + dh, ww = wcol + dw;
      if (hh >= 0 && hh < HH && ww >= 0 && ww < WW)
        acc += w9[c * 9 + (dh + 1) * 3 + (dw + 1)] * (float)p[hh * WW + ww];
    }
  dst[((size_t)b * C + c) * HW + s] = (h16)acc;

  if (sumsq != nullptr) {
    __shared__ float red[256];
    red[threadIdx.x] = acc * acc;
    __syncthreads();
    for (int st = 128; st > 0; st >>= 1) {
      if (threadIdx.x < st) red[threadIdx.x] += red[threadIdx.x + st];
      __syncthreads();
    }
    if (threadIdx.x == 0) atomicAdd(&sumsq[b * C + c], red[0]);
  }
}

// ---------------- Kernel 3: S += Q K^T (K-split WMMA GEMM, LDS-staged K tile) --------
#define KSPLIT 64
__global__ void __launch_bounds__(256) qk_gemm_kernel(
    const h16* __restrict__ q, const h16* __restrict__ k,
    float* __restrict__ S)
{
  __shared__ h16 kt[C * 32];                   // 8 KB: k[d=0..127][kk..kk+31]
  const int tid  = threadIdx.x;
  const int lane = tid & 31;
  const int mt   = tid >> 5;                   // wave id = m tile
  const int l16  = lane & 15;
  const int half = lane >> 4;
  const int b    = blockIdx.x >> 6;
  const int ks   = (blockIdx.x & 63) * (HW / KSPLIT);
  const h16* qb = q + (size_t)b * C * HW;
  const h16* kp = k + (size_t)b * C * HW;

  v8f acc[8];
#pragma unroll
  for (int t = 0; t < 8; ++t) acc[t] = (v8f){};

  const int srow = tid >> 1;                   // staging row (channel d)
  const int scol = (tid & 1) * 16;             // staging col within 32-k chunk

  for (int kk = ks; kk < ks + HW / KSPLIT; kk += 32) {
    // stage k tile cooperatively (each thread 32B via two async b128)
    {
      const unsigned dst = lds_off_of(&kt[srow * 32 + scol]);
      const h16* src = kp + (size_t)srow * HW + kk + scol;
      async_g2l_b128(dst, src);
      async_g2l_b128(dst + 16u, src + 8);
      wait_async0();
    }
    __syncthreads();

    v16h afrag;
#pragma unroll
    for (int e = 0; e < 16; ++e)               // A: [M=qchan][K=spatial] (global)
      afrag[e] = qb[(size_t)(mt * 16 + l16) * HW + kk + kmap(e, half)];
#pragma unroll
    for (int nt = 0; nt < 8; ++nt) {
      v16h bfrag;
#pragma unroll
      for (int e = 0; e < 16; ++e)             // B: [K=spatial][N=kchan] (LDS)
        bfrag[e] = kt[(nt * 16 + l16) * 32 + kmap(e, half)];
      acc[nt] = __builtin_amdgcn_wmma_f32_16x16x32_f16(false, afrag, false, bfrag, (short)0, acc[nt], false, false);
    }
    __syncthreads();                           // before next stage overwrites tile
  }
#pragma unroll
  for (int nt = 0; nt < 8; ++nt)
#pragma unroll
    for (int i = 0; i < 8; ++i) {
      const int m = mt * 16 + i + half * 8;
      const int nn = nt * 16 + l16;
      atomicAdd(&S[((size_t)b * C + m) * C + nn], acc[nt][i]);
    }
}

// ---------------- Kernel 4: L2-norm scaling + row softmax ----------------
__global__ void __launch_bounds__(256) softmax_kernel(
    const float* __restrict__ S, const float* __restrict__ ssq,
    const float* __restrict__ ssk, float* __restrict__ A)
{
  const int lane = threadIdx.x & 31;
  const int wid  = threadIdx.x >> 5;
  const int row  = blockIdx.x * 8 + wid;      // [0, 16*128)
  const int b = row >> 7, c = row & 127;
  const float invq = 1.f / fmaxf(sqrtf(ssq[b * C + c]), 1e-12f);
  const float* srow = S + ((size_t)b * C + c) * C;

  float v[4];
  float mx = -3.4e38f;
#pragma unroll
  for (int t = 0; t < 4; ++t) {
    const int d = lane + 32 * t;
    const float invk = 1.f / fmaxf(sqrtf(ssk[b * C + d]), 1e-12f);
    v[t] = srow[d] * invq * invk;
    mx = fmaxf(mx, v[t]);
  }
  for (int off = 16; off > 0; off >>= 1) mx = fmaxf(mx, __shfl_xor(mx, off, 32));
  float sum = 0.f;
#pragma unroll
  for (int t = 0; t < 4; ++t) { v[t] = __expf(v[t] - mx); sum += v[t]; }
  for (int off = 16; off > 0; off >>= 1) sum += __shfl_xor(sum, off, 32);
  const float inv = 1.f / sum;
  float* arow = A + ((size_t)b * C + c) * C;
#pragma unroll
  for (int t = 0; t < 4; ++t) arow[lane + 32 * t] = v[t] * inv;
}

// ---------------- Kernel 5: M = Wa * A (tiny per-batch GEMM, f32 FMA) ----------------
__global__ void __launch_bounds__(256) wa_a_kernel(
    const float* __restrict__ aw, const float* __restrict__ A,
    h16* __restrict__ M)
{
  const int idx = blockIdx.x * 256 + threadIdx.x;   // [0, 16*128*128)
  const int d = idx & 127;
  const int m = (idx >> 7) & 127;
  const int b = idx >> 14;
  const float* ab_ = A + (size_t)b * C * C;
  float s = 0.f;
#pragma unroll 4
  for (int cc = 0; cc < C; ++cc) s += aw[m * C + cc] * ab_[cc * C + d];
  M[idx] = (h16)s;
}

// ---------------- Kernel 6: out = (Wa*A) * V + ab (WMMA, LDS-staged V tile) ----------
__global__ void __launch_bounds__(256) av_out_kernel(
    const h16* __restrict__ M, const h16* __restrict__ v,
    const float* __restrict__ abias, float* __restrict__ out)
{
  __shared__ h16 vs[C * 16];                   // 4 KB: v[d=0..127][n0..n0+15]
  const int tid  = threadIdx.x;
  const int lane = tid & 31;
  const int mt   = tid >> 5;                   // wave id = out-channel tile
  const int l16  = lane & 15;
  const int half = lane >> 4;
  const int b    = blockIdx.x >> 10;
  const int nt   = blockIdx.x & 1023;
  const int n0   = nt * 16;
  const h16* mb = M + (size_t)b * C * C;
  const h16* vp = v + (size_t)b * C * HW;

  // stage v tile (each thread 16B via one async b128)
  {
    const int row = tid >> 1;
    const int col = (tid & 1) * 8;
    const unsigned dst = lds_off_of(&vs[row * 16 + col]);
    const h16* src = vp + (size_t)row * HW + n0 + col;
    async_g2l_b128(dst, src);
    wait_async0();
  }
  __syncthreads();

  v8f acc = {};
  for (int kb = 0; kb < C; kb += 32) {
    v16h afrag, bfrag;
#pragma unroll
    for (int e = 0; e < 16; ++e) {
      const int kk = kb + kmap(e, half);
      afrag[e] = mb[(mt * 16 + l16) * C + kk]; // A: [M=cout][K=d] (global, tiny)
      bfrag[e] = vs[kk * 16 + l16];            // B: [K=d][N=spatial] (LDS)
    }
    acc = __builtin_amdgcn_wmma_f32_16x16x32_f16(false, afrag, false, bfrag, (short)0, acc, false, false);
  }
#pragma unroll
  for (int i = 0; i < 8; ++i) {
    const int m = mt * 16 + i + half * 8;
    out[((size_t)b * C + m) * HW + n0 + l16] = acc[i] + abias[m];
  }
}

extern "C" void kernel_launch(void* const* d_in, const int* in_sizes, int n_in,
                              void* d_out, int out_size, void* d_ws, size_t ws_size,
                              hipStream_t stream) {
  const float* x   = (const float*)d_in[0];
  const float* qw  = (const float*)d_in[1];
  const float* qb  = (const float*)d_in[2];
  const float* qdw = (const float*)d_in[3];
  const float* qdb = (const float*)d_in[4];
  const float* kw  = (const float*)d_in[5];
  const float* kb  = (const float*)d_in[6];
  const float* kdw = (const float*)d_in[7];
  const float* kdb = (const float*)d_in[8];
  const float* vw  = (const float*)d_in[9];
  const float* vb  = (const float*)d_in[10];
  const float* vdw = (const float*)d_in[11];
  const float* vdb = (const float*)d_in[12];
  const float* aw  = (const float*)d_in[13];
  const float* ab  = (const float*)d_in[14];
  float* out = (float*)d_out;

  // Workspace carve-up
  char* ws = (char*)d_ws;
  const size_t tensN = (size_t)NB * C * HW;          // 33,554,432 elements
  h16* q0 = (h16*)ws;            ws += tensN * sizeof(h16);
  h16* k0 = (h16*)ws;            ws += tensN * sizeof(h16);
  h16* v0 = (h16*)ws;            ws += tensN * sizeof(h16);
  h16* qd = (h16*)ws;            ws += tensN * sizeof(h16);
  h16* kd = (h16*)ws;            ws += tensN * sizeof(h16);
  h16* vd = (h16*)ws;            ws += tensN * sizeof(h16);
  float* ssq = (float*)ws;       ws += (size_t)NB * C * sizeof(float);
  float* ssk = (float*)ws;       ws += (size_t)NB * C * sizeof(float);
  float* S   = (float*)ws;       ws += (size_t)NB * C * C * sizeof(float);
  float* A   = (float*)ws;       ws += (size_t)NB * C * C * sizeof(float);
  h16*   Mm  = (h16*)ws;         ws += (size_t)NB * C * C * sizeof(h16);
  h16*   w3  = (h16*)ws;         ws += (size_t)3 * C * C * sizeof(h16);

  // 0) zero accumulation buffers (ssq, ssk, S contiguous) + convert weights to f16
  {
    const int nz = NB * C + NB * C + NB * C * C;     // 266,240 floats
    zero_f32_kernel<<<(nz + 255) / 256, 256, 0, stream>>>(ssq, nz);
    cvt_weights_kernel<<<3 * C * C / 256, 256, 0, stream>>>(qw, kw, vw, w3);
  }

  // 1) fused QKV projections (WMMA, async-staged + LDS-transposed x tile)
  qkv_proj_kernel<<<NB * 1024, 256, 0, stream>>>(
      x, w3, qb, kb, vb, q0, k0, v0);

  // 2) depthwise 3x3 + bias (+ sumsq for q,k)
  const int dwGrid = NB * C * (HW / 256);
  dwconv_kernel<<<dwGrid, 256, 0, stream>>>(q0, qdw, qdb, qd, ssq);
  dwconv_kernel<<<dwGrid, 256, 0, stream>>>(k0, kdw, kdb, kd, ssk);
  dwconv_kernel<<<dwGrid, 256, 0, stream>>>(v0, vdw, vdb, vd, nullptr);

  // 3) S = Q K^T, K-split with atomic accumulation
  qk_gemm_kernel<<<NB * KSPLIT, 256, 0, stream>>>(qd, kd, S);

  // 4) normalize + softmax
  softmax_kernel<<<NB * C / 8, 256, 0, stream>>>(S, ssq, ssk, A);

  // 5) Mm = Wa * A  (folds output 1x1 conv into the attention matrix)
  wa_a_kernel<<<NB * C * C / 256, 256, 0, stream>>>(aw, A, Mm);

  // 6) out = Mm * V + ab (WMMA, LDS-staged v tile): one block per (b, ntile)
  av_out_kernel<<<NB * 1024, 256, 0, stream>>>(Mm, vd, ab, out);
}
